// RelativePositionAttention_84207128805914
// MI455X (gfx1250) — matrix operand order repaired and checked by reading, not compiled, not run on hardware
//
#include <hip/hip_runtime.h>
#include <hip/hip_bf16.h>

typedef __attribute__((ext_vector_type(16))) _Float16 v16h;
typedef __attribute__((ext_vector_type(8)))  _Float16 v8h;
typedef __attribute__((ext_vector_type(8)))  float    v8f;

// ---------------------------------------------------------------------------
// Problem constants
// ---------------------------------------------------------------------------
#define BB   8
#define HS   32
#define WS   32
#define DIM  768
#define NH   12
#define HD   64
#define LL   1024            // HS*WS
#define MROWS (BB*LL)        // 8192
#define BH   (BB*NH)         // 96

// ---------------------------------------------------------------------------
// WMMA helpers (CDNA5 v_wmma_f32_16x16x32_f16, layouts per ISA 7.12.2)
// ---------------------------------------------------------------------------
__device__ __forceinline__ v8f wmma_f16(v16h a, v16h b, v8f c) {
    return __builtin_amdgcn_wmma_f32_16x16x32_f16(
        /*neg_a=*/false, a, /*neg_b=*/false, b,
        /*c_mod=*/(short)0, c, /*reuse_a=*/false, /*reuse_b=*/false);
}

// A fragment: 16(M) x 32(K), f16 row-major source with row stride ld (elems).
// lane: row = lane&15; klo = 8*(lane>=16); halves 0..7 -> K=klo..klo+7,
// halves 8..15 -> K=klo+16..klo+23   (two 16B loads)
__device__ __forceinline__ v16h load_a16(const _Float16* p, int ld, int lane) {
    int m   = lane & 15;
    int klo = (lane >> 4) << 3;
    const _Float16* q = p + (size_t)m * ld + klo;
    union { v16h v; v8h h[2]; } u;
    u.h[0] = *(const v8h*)(q);
    u.h[1] = *(const v8h*)(q + 16);
    return u.v;
}

// B fragment: 32(K) x 16(N). Source layout: row n (output col), contiguous K,
// row stride ld. lane: col = lane&15; klo = 8*(lane>=16); halves 0..7 -> K=klo..klo+7,
// halves 8..15 -> K=klo+16..klo+23   (two 16B loads)
__device__ __forceinline__ v16h load_b16(const _Float16* p, int ld, int lane) {
    int n   = lane & 15;
    int klo = (lane >> 4) << 3;
    const _Float16* q = p + (size_t)n * ld + klo;
    union { v16h v; v8h h[2]; } u;
    u.h[0] = *(const v8h*)(q);
    u.h[1] = *(const v8h*)(q + 16);
    return u.v;
}

// ---------------------------------------------------------------------------
// Pass 0: fp32 -> f16 conversion
// ---------------------------------------------------------------------------
__global__ void cvt_f32_f16(const float* __restrict__ in,
                            _Float16* __restrict__ out, int n) {
    int i = blockIdx.x * 256 + threadIdx.x;
    if (i < n) out[i] = (_Float16)in[i];
}

// ---------------------------------------------------------------------------
// Pass 1: QKV GEMM.  C[8192][2304] = X[8192][768] * W[2304][768]^T + b
// Scatter: Q,K -> [bh][l][64] f16 ; V -> transposed [bh][64][l] f16
// Block: 256 thr = 8 waves; block tile 128x128; wave tile 32(M)x64(N)
// ---------------------------------------------------------------------------
__global__ __launch_bounds__(256, 1)
void gemm_qkv(const _Float16* __restrict__ X, const _Float16* __restrict__ W,
              const float* __restrict__ bias,
              _Float16* __restrict__ Qh, _Float16* __restrict__ Kh,
              _Float16* __restrict__ Vt) {
    int lane = threadIdx.x & 31, wv = threadIdx.x >> 5;
    int wm = wv >> 1, wn = wv & 1;
    int mBase = blockIdx.y * 128 + wm * 32;
    int nBase = blockIdx.x * 128 + wn * 64;

    v8f acc[2][4] = {};
    for (int k = 0; k < DIM; k += 32) {
        v16h a0 = load_a16(X + (size_t)mBase * DIM + k, DIM, lane);
        v16h a1 = load_a16(X + (size_t)(mBase + 16) * DIM + k, DIM, lane);
#pragma unroll
        for (int bf = 0; bf < 4; ++bf) {
            v16h bfr = load_b16(W + (size_t)(nBase + bf * 16) * DIM + k, DIM, lane);
            acc[0][bf] = wmma_f16(a0, bfr, acc[0][bf]);
            acc[1][bf] = wmma_f16(a1, bfr, acc[1][bf]);
        }
    }

    int rlo = (lane >> 4) << 3, cc = lane & 15;
#pragma unroll
    for (int af = 0; af < 2; ++af)
#pragma unroll
        for (int bf = 0; bf < 4; ++bf)
#pragma unroll
            for (int r = 0; r < 8; ++r) {
                int m = mBase + af * 16 + rlo + r;
                int n = nBase + bf * 16 + cc;
                float v = acc[af][bf][r] + bias[n];
                int which = n / DIM, rem = n % DIM;
                int head = rem >> 6, d = rem & 63;
                int b = m >> 10, l = m & 1023;
                int bh = b * NH + head;
                _Float16 hv = (_Float16)v;
                if (which == 0)      Qh[((size_t)bh * LL + l) * HD + d] = hv;
                else if (which == 1) Kh[((size_t)bh * LL + l) * HD + d] = hv;
                else                 Vt[((size_t)bh * HD + d) * LL + l] = hv;
            }
}

// ---------------------------------------------------------------------------
// Pass 2: decomposed rel-pos bias tables.
// Rh[bh][qrow][kh] = sum_c q[c] * rel_pos_h[h-kh+31][c]   (h = qrow/32)
// Rw[bh][qrow][kw] = sum_c q[c] * rel_pos_w[w-kw+31][c]   (w = qrow%32)
// ---------------------------------------------------------------------------
__global__ void rel_bias(const _Float16* __restrict__ Qh,
                         const float* __restrict__ rph,
                         const float* __restrict__ rpw,
                         float* __restrict__ Rh, float* __restrict__ Rw) {
    size_t idx = (size_t)blockIdx.x * 256 + threadIdx.x;   // bh*32768 + qrow*32 + j
    int j    = (int)(idx & 31);
    int qrow = (int)((idx >> 5) & 1023);
    int bh   = (int)(idx >> 15);
    if (bh >= BH) return;
    int h = qrow >> 5, w = qrow & 31;
    const _Float16* q  = Qh + ((size_t)bh * LL + qrow) * HD;
    const float*    ph = rph + (size_t)(h - j + 31) * HD;
    const float*    pw = rpw + (size_t)(w - j + 31) * HD;
    float sh = 0.f, sw = 0.f;
#pragma unroll 8
    for (int c = 0; c < HD; ++c) {
        float qc = (float)q[c];
        sh += qc * ph[c];
        sw += qc * pw[c];
    }
    Rh[idx] = sh;
    Rw[idx] = sw;
}

// ---------------------------------------------------------------------------
// Pass 3: flash attention with fused rel-pos bias.
// grid = (8 q-tiles of 128, 96 bh). 8 waves/block, one 16-row q-tile per wave.
// Key blocks of 32 (aligned to W=32 so kh is a per-block scalar, kw per lane
// is loop-invariant). P re-layout via 1KB wave-private LDS bounce.
// ---------------------------------------------------------------------------
__global__ __launch_bounds__(256, 1)
void attn_kernel(const _Float16* __restrict__ Qh, const _Float16* __restrict__ Kh,
                 const _Float16* __restrict__ Vt, const float* __restrict__ Rh,
                 const float* __restrict__ Rw, _Float16* __restrict__ Oh) {
    __shared__ _Float16 pbuf[8][16 * 32];
    int lane = threadIdx.x & 31, wv = threadIdx.x >> 5;
    int bh = blockIdx.y;
    int qrow0 = blockIdx.x * 128 + wv * 16;

    const _Float16* Q = Qh + (size_t)bh * LL * HD;
    const _Float16* K = Kh + (size_t)bh * LL * HD;
    const _Float16* V = Vt + (size_t)bh * LL * HD;
    const float* RhB  = Rh + (size_t)bh * LL * 32;
    const float* RwB  = Rw + (size_t)bh * LL * 32;

    int rlo = (lane >> 4) << 3, cc = lane & 15;

    v16h a0 = load_a16(Q + (size_t)qrow0 * HD, HD, lane);        // d 0..31
    v16h a1 = load_a16(Q + (size_t)qrow0 * HD + 32, HD, lane);   // d 32..63

    float rw0[8], rw1[8], mr[8], lr[8];
#pragma unroll
    for (int r = 0; r < 8; ++r) {
        int qr = qrow0 + rlo + r;
        rw0[r] = RwB[(size_t)qr * 32 + cc];
        rw1[r] = RwB[(size_t)qr * 32 + 16 + cc];
        mr[r] = -1e30f;
        lr[r] = 0.f;
    }
    v8f o[4] = {};

    for (int kb = 0; kb < 32; ++kb) {
        int n0 = kb * 32;
        v16h b00 = load_b16(K + (size_t)n0 * HD, HD, lane);
        v16h b01 = load_b16(K + (size_t)n0 * HD + 32, HD, lane);
        v16h b10 = load_b16(K + (size_t)(n0 + 16) * HD, HD, lane);
        v16h b11 = load_b16(K + (size_t)(n0 + 16) * HD + 32, HD, lane);

        v8f s0 = {}; s0 = wmma_f16(a0, b00, s0); s0 = wmma_f16(a1, b01, s0);
        v8f s1 = {}; s1 = wmma_f16(a0, b10, s1); s1 = wmma_f16(a1, b11, s1);

        float p0[8], p1[8], t[8];
#pragma unroll
        for (int r = 0; r < 8; ++r) {
            int qr = qrow0 + rlo + r;
            float bias_h = RhB[(size_t)qr * 32 + kb];    // kh == kb for all 32 keys
            p0[r] = s0[r] * 0.125f + bias_h + rw0[r];
            p1[r] = s1[r] * 0.125f + bias_h + rw1[r];
            t[r] = fmaxf(p0[r], p1[r]);
        }
#pragma unroll
        for (int mask = 1; mask <= 8; mask <<= 1)
#pragma unroll
            for (int r = 0; r < 8; ++r)
                t[r] = fmaxf(t[r], __shfl_xor(t[r], mask, 32));

        float alpha[8];
#pragma unroll
        for (int r = 0; r < 8; ++r) {
            float mn = fmaxf(mr[r], t[r]);
            alpha[r] = __expf(mr[r] - mn);
            mr[r] = mn;
            p0[r] = __expf(p0[r] - mn);
            p1[r] = __expf(p1[r] - mn);
            t[r] = p0[r] + p1[r];
        }
#pragma unroll
        for (int mask = 1; mask <= 8; mask <<= 1)
#pragma unroll
            for (int r = 0; r < 8; ++r)
                t[r] += __shfl_xor(t[r], mask, 32);
#pragma unroll
        for (int r = 0; r < 8; ++r) lr[r] = lr[r] * alpha[r] + t[r];
#pragma unroll
        for (int dg = 0; dg < 4; ++dg)
#pragma unroll
            for (int r = 0; r < 8; ++r) o[dg][r] *= alpha[r];

        // P (16x32, C-layout) -> LDS -> A-fragment layout
        _Float16* pb = pbuf[wv];
#pragma unroll
        for (int r = 0; r < 8; ++r) {
            int ml = rlo + r;
            pb[ml * 32 + cc]      = (_Float16)p0[r];
            pb[ml * 32 + 16 + cc] = (_Float16)p1[r];
        }
        asm volatile("s_wait_dscnt 0" ::: "memory");
        v16h ap = load_a16(pb, 32, lane);
#pragma unroll
        for (int dg = 0; dg < 4; ++dg) {
            v16h bv = load_b16(V + (size_t)(dg * 16) * LL + n0, LL, lane);
            o[dg] = wmma_f16(ap, bv, o[dg]);
        }
    }

    int b = bh / NH, head = bh % NH;
#pragma unroll
    for (int dg = 0; dg < 4; ++dg)
#pragma unroll
        for (int r = 0; r < 8; ++r) {
            int qr = qrow0 + rlo + r;
            float val = o[dg][r] / lr[r];
            Oh[((size_t)b * LL + qr) * DIM + head * HD + dg * 16 + cc] =
                (_Float16)val;
        }
}

// ---------------------------------------------------------------------------
// Pass 4: output projection. out[8192][768] = O[8192][768]*Wp[768][768]^T + b
// ---------------------------------------------------------------------------
__global__ __launch_bounds__(256, 1)
void gemm_proj(const _Float16* __restrict__ A, const _Float16* __restrict__ W,
               const float* __restrict__ bias, float* __restrict__ out) {
    int lane = threadIdx.x & 31, wv = threadIdx.x >> 5;
    int wm = wv >> 1, wn = wv & 1;
    int mBase = blockIdx.y * 128 + wm * 32;
    int nBase = blockIdx.x * 128 + wn * 64;

    v8f acc[2][4] = {};
    for (int k = 0; k < DIM; k += 32) {
        v16h a0 = load_a16(A + (size_t)mBase * DIM + k, DIM, lane);
        v16h a1 = load_a16(A + (size_t)(mBase + 16) * DIM + k, DIM, lane);
#pragma unroll
        for (int bf = 0; bf < 4; ++bf) {
            v16h bfr = load_b16(W + (size_t)(nBase + bf * 16) * DIM + k, DIM, lane);
            acc[0][bf] = wmma_f16(a0, bfr, acc[0][bf]);
            acc[1][bf] = wmma_f16(a1, bfr, acc[1][bf]);
        }
    }

    int rlo = (lane >> 4) << 3, cc = lane & 15;
#pragma unroll
    for (int af = 0; af < 2; ++af)
#pragma unroll
        for (int bf = 0; bf < 4; ++bf)
#pragma unroll
            for (int r = 0; r < 8; ++r) {
                int m = mBase + af * 16 + rlo + r;
                int n = nBase + bf * 16 + cc;
                out[(size_t)m * DIM + n] = acc[af][bf][r] + bias[n];
            }
}

// ---------------------------------------------------------------------------
// Launcher
// ---------------------------------------------------------------------------
extern "C" void kernel_launch(void* const* d_in, const int* in_sizes, int n_in,
                              void* d_out, int out_size, void* d_ws, size_t ws_size,
                              hipStream_t stream) {
    const float* x      = (const float*)d_in[0];   // 8*32*32*768
    const float* qkv_w  = (const float*)d_in[1];   // 2304*768
    const float* qkv_b  = (const float*)d_in[2];   // 2304
    const float* proj_w = (const float*)d_in[3];   // 768*768
    const float* proj_b = (const float*)d_in[4];   // 768
    const float* rph    = (const float*)d_in[5];   // 63*64
    const float* rpw    = (const float*)d_in[6];   // 63*64
    float* out = (float*)d_out;

    // workspace layout (bytes)
    char* w = (char*)d_ws;
    constexpr size_t SZ_XH  = (size_t)MROWS * DIM * 2;       // 12.6 MB
    constexpr size_t SZ_WQ  = (size_t)3 * DIM * DIM * 2;     //  3.5 MB
    constexpr size_t SZ_WP  = (size_t)DIM * DIM * 2;         //  1.2 MB
    constexpr size_t SZ_QKV = (size_t)BH * LL * HD * 2;      // 12.6 MB each
    constexpr size_t SZ_R   = (size_t)BH * LL * 32 * 4;      // 12.6 MB each
    constexpr size_t SZ_OH  = (size_t)MROWS * DIM * 2;       // 12.6 MB

    size_t off = 0;
    _Float16* Xh  = (_Float16*)(w + off); off += SZ_XH;
    _Float16* Wq  = (_Float16*)(w + off); off += SZ_WQ;
    _Float16* Wp  = (_Float16*)(w + off); off += SZ_WP;
    _Float16* Qh  = (_Float16*)(w + off); off += SZ_QKV;
    _Float16* Kh  = (_Float16*)(w + off); off += SZ_QKV;
    _Float16* Vt  = (_Float16*)(w + off); off += SZ_QKV;
    float*    Rh  = (float*)(w + off);    off += SZ_R;
    float*    Rw  = (float*)(w + off);    off += SZ_R;
    _Float16* Oh  = (_Float16*)(w + off); off += SZ_OH;
    (void)ws_size; (void)in_sizes; (void)n_in; (void)out_size;

    // Pass 0: conversions
    {
        int n0 = MROWS * DIM;
        cvt_f32_f16<<<(n0 + 255) / 256, 256, 0, stream>>>(x, Xh, n0);
        int n1 = 3 * DIM * DIM;
        cvt_f32_f16<<<(n1 + 255) / 256, 256, 0, stream>>>(qkv_w, Wq, n1);
        int n2 = DIM * DIM;
        cvt_f32_f16<<<(n2 + 255) / 256, 256, 0, stream>>>(proj_w, Wp, n2);
    }

    // Pass 1: QKV GEMM + scatter
    gemm_qkv<<<dim3((3 * DIM) / 128, MROWS / 128), 256, 0, stream>>>(
        Xh, Wq, qkv_b, Qh, Kh, Vt);

    // Pass 2: rel-pos bias tables
    {
        size_t total = (size_t)BH * LL * 32;
        rel_bias<<<(unsigned)((total + 255) / 256), 256, 0, stream>>>(
            Qh, rph, rpw, Rh, Rw);
    }

    // Pass 3: flash attention
    attn_kernel<<<dim3(LL / 128, BH), 256, 0, stream>>>(Qh, Kh, Vt, Rh, Rw, Oh);

    // Pass 4: projection
    gemm_proj<<<dim3(DIM / 128, MROWS / 128), 256, 0, stream>>>(
        Oh, Wp, proj_b, out);
}
